// Codec_53807350284793
// MI455X (gfx1250) — matrix operand order repaired and checked by reading, not compiled, not run on hardware
//
#include <hip/hip_runtime.h>
#include <hip/hip_bf16.h>
#include <math.h>

// ---------------------------------------------------------------------------
// Wavefront recurrent-MLP codec for MI455X (gfx1250, wave32, WMMA).
//   t = 2r + k wavefront; 1 persistent workgroup of 8 waves; per-wave 16-cell
//   batches evaluated with v_wmma_f32_16x16x32_f16 against f16 weights in LDS.
//   All layer shapes are compile-time specialized -> straight-line WMMA code.
// ---------------------------------------------------------------------------

typedef __attribute__((ext_vector_type(16))) _Float16 v16h;
typedef __attribute__((ext_vector_type(8)))  _Float16 v8h;
typedef __attribute__((ext_vector_type(8)))  float    v8f;

#define IMG_H   256
#define IMG_W   256
#define OROWS   253            // row 0 is the zero "prev" row
#define OSTRIDE (256*12)       // floats per output-grid row
#define TMAX    753            // 2*251 + 251

// Wt (f16, halves) offsets per (mlp, layer); layout Wt[Npad][Kpad]
constexpr int WOFFc[3][8] = {
  {0, 1024, 2048, 3072, 4096, 5120, 6144, 7168},
  {7680, 9216, 12288, 15360, 18432, 21504, 24576, 27648},
  {28672, 29696, 30720, 31744, 32768, 33792, 34816, 35840}};
constexpr int BOFFc[3][8] = {
  {0, 32, 64, 96, 128, 160, 192, 224},
  {240, 288, 336, 384, 432, 480, 528, 576},
  {592, 624, 656, 688, 720, 752, 784, 816}};
static const int hDIN[3][8]  = {
  {28,24,24,24,24,24,24,24}, {28,48,48,48,48,48,48,48}, {28,24,24,24,24,24,24,24}};
static const int hDOUT[3][8] = {
  {24,24,24,24,24,24,24,4},  {48,48,48,48,48,48,48,4},  {24,24,24,24,24,24,24,4}};

#define WT_HALVES   36352
#define WT_BYTES    (WT_HALVES*2)     // 72704
#define BIAS_FLOATS 832
#define BIAS_OFF    WT_BYTES          // byte offset in ws / LDS
#define OGRID_OFF   (WT_BYTES + 3328) // 76032 bytes, 16B aligned
#define SMEM_BYTES  (OGRID_OFF + 8*16*64*2)  // + act scratch = 92416

// neighbor offsets (dy in {-2,-1} x dx in {-2..2}) + (0,-2),(0,-1)
__device__ __constant__ int cDY[12] = {-2,-2,-2,-2,-2,-1,-1,-1,-1,-1, 0, 0};
__device__ __constant__ int cDX[12] = {-2,-1, 0, 1, 2,-2,-1, 0, 1, 2,-2,-1};

// ---- WMMA fragment helpers (ISA 7.12.2 16-bit layouts, wave32) ------------
__device__ __forceinline__ v16h load_a_frag(const _Float16* act, int lane, int kt) {
  // A 16x32 f16: lanes 0-15 row m hold K {0..7,16..23}; lanes 16-31 K {8..15,24..31}
  const int m  = lane & 15;
  const int hi = lane >> 4;
  const _Float16* p = act + m * 64 + kt * 32 + hi * 8;
  v8h lo = *(const v8h*)(p);
  v8h hh = *(const v8h*)(p + 16);
  return __builtin_shufflevector(lo, hh, 0,1,2,3,4,5,6,7,8,9,10,11,12,13,14,15);
}
template<int KPAD>
__device__ __forceinline__ v16h load_b_frag(const _Float16* wt, int lane,
                                            int nt, int kt) {
  // B 32x16 f16: lanes 0-15 col n hold K {0..7,16..23}; lanes 16-31 K {8..15,24..31}
  const int n  = nt * 16 + (lane & 15);
  const int hi = lane >> 4;
  const _Float16* p = wt + n * KPAD + kt * 32 + hi * 8;
  v8h lo = *(const v8h*)(p);
  v8h hh = *(const v8h*)(p + 16);
  return __builtin_shufflevector(lo, hh, 0,1,2,3,4,5,6,7,8,9,10,11,12,13,14,15);
}
__device__ __forceinline__ v8f wmma_f16(v16h a, v16h b, v8f c) {
  return __builtin_amdgcn_wmma_f32_16x16x32_f16(false, a, false, b, (short)0, c,
                                                false, false);
}

// ---- one fully specialized hidden layer (bias + leaky_relu 0.01) ----------
template<int KPAD, int KT, int NT>
__device__ __forceinline__ void hidden_layer(const _Float16* __restrict__ wt,
                                             const float* __restrict__ bl,
                                             _Float16* act,
                                             int lane, int m16, int hi) {
  v8f acc[NT];
#pragma unroll
  for (int nt = 0; nt < NT; ++nt)
#pragma unroll
    for (int j = 0; j < 8; ++j) acc[nt][j] = 0.0f;

#pragma unroll
  for (int kt = 0; kt < KT; ++kt) {
    v16h af = load_a_frag(act, lane, kt);
#pragma unroll
    for (int nt = 0; nt < NT; ++nt) {
      v16h bf = load_b_frag<KPAD>(wt, lane, nt, kt);
      acc[nt] = wmma_f16(af, bf, acc[nt]);
    }
  }

#pragma unroll
  for (int nt = 0; nt < NT; ++nt) {
    const int n = nt * 16 + m16;
    const float bn = bl[n];
#pragma unroll
    for (int j = 0; j < 8; ++j) {
      const int mr = j + 8 * hi;
      float vv = acc[nt][j] + bn;
      vv = (vv > 0.0f) ? vv : 0.01f * vv;
      act[mr * 64 + n] = (_Float16)vv;
    }
  }
  asm volatile("s_wait_dscnt 0x0" ::: "memory");
}

// ---- final layer accumulator (single 16-wide N tile, cols 0..3 valid) -----
template<int KPAD, int KT>
__device__ __forceinline__ v8f final_acc(const _Float16* __restrict__ wt,
                                         _Float16* act, int lane) {
  v8f acc;
#pragma unroll
  for (int j = 0; j < 8; ++j) acc[j] = 0.0f;
#pragma unroll
  for (int kt = 0; kt < KT; ++kt) {
    v16h af = load_a_frag(act, lane, kt);
    v16h bf = load_b_frag<KPAD>(wt, lane, 0, kt);
    acc = wmma_f16(af, bf, acc);
  }
  return acc;
}

// ---- one full MLP (feature fill + 8 layers + scatter) ---------------------
template<int MLP, int NCH>
__device__ __forceinline__ void run_mlp(const float* __restrict__ Xc,
                                        float* O,
                                        const _Float16* sWt, const float* sB,
                                        _Float16* act,
                                        int t, int rmin, int count,
                                        int wave, int lane, int m16, int hi) {
  // ---------- feature fill into act[16][64] (f16), cols>=28 zeroed ---------
  for (int idx = lane; idx < 16 * 64; idx += 32) {
    const int m = idx >> 6;
    const int j = idx & 63;
    float v = 0.0f;
    if (j < 28) {
      int cell = wave * 16 + m;
      int r = rmin + cell; if (r > 251) r = 251;
      int k = t - 2 * r;   if (k < 0) k = 0; if (k > 251) k = 251;
      const int rr = r + 2, cc = k + 2;
      if (j < 12) {
        v = Xc[(rr + cDY[j]) * IMG_W + (cc + cDX[j])];
      } else {
        const int jj = j - 12, ch4 = jj >> 2, pos = jj & 3;
        const int orow = (pos < 3) ? r : (r + 1);      // prev row r-1 -> idx r
        const int ocol = (pos < 3) ? (cc - 1 + pos) : (cc - 1);
        v = O[orow * OSTRIDE + ocol * 12 + MLP * 4 + ch4];
      }
    }
    act[m * 64 + j] = (_Float16)v;
  }
  asm volatile("s_wait_dscnt 0x0" ::: "memory");

  constexpr int KPAD = (NCH == 48) ? 64 : 32;
  constexpr int KT   = (NCH == 48) ? 2 : 1;
  constexpr int NT   = (NCH == 48) ? 3 : 2;

  // layer 0 always has K = 28 -> KPAD 32, 1 K-tile
  hidden_layer<32, 1, NT>(sWt + WOFFc[MLP][0], sB + BOFFc[MLP][0],
                          act, lane, m16, hi);
#pragma unroll
  for (int l = 1; l < 7; ++l)
    hidden_layer<KPAD, KT, NT>(sWt + WOFFc[MLP][l], sB + BOFFc[MLP][l],
                               act, lane, m16, hi);

  // final layer: clip, fmod-wrap residual, scatter into O grid
  v8f acc = final_acc<KPAD, KT>(sWt + WOFFc[MLP][7], act, lane);
  const float* bl = sB + BOFFc[MLP][7];
  if (m16 < 4) {
    const int n = m16;
    const float bn = bl[n];
#pragma unroll
    for (int j = 0; j < 8; ++j) {
      const int mr   = j + 8 * hi;
      const int cell = wave * 16 + mr;
      if (cell < count) {
        const int r  = rmin + cell;
        const int k  = t - 2 * r;
        const int cc = k + 2;
        float vv = acc[j] + bn;
        vv = fminf(1.0f, fmaxf(-1.0f, vv));
        if (n == 3) {
          const float cen = Xc[(r + 2) * IMG_W + cc];
          vv = fmodf(cen - vv + 1.0f, 2.0f) - 1.0f;
        }
        O[(r + 1) * OSTRIDE + cc * 12 + MLP * 4 + n] = vv;
      }
    }
  }
}

// ---- weight prep: W[k][n] f32 -> Wt[n_pad][k_pad] f16 (zero padded) -------
__global__ void prep_weights_kernel(const float* __restrict__ Wsrc,
                                    const float* __restrict__ bsrc,
                                    _Float16* __restrict__ wt,
                                    float* __restrict__ bout,
                                    int d0, int d1, int kpad, int npad) {
  const int total  = npad * kpad;
  const int stride = gridDim.x * blockDim.x;
  for (int i = blockIdx.x * blockDim.x + threadIdx.x; i < total; i += stride) {
    int n = i / kpad, k = i - n * kpad;
    float v = (n < d1 && k < d0) ? Wsrc[k * d1 + n] : 0.0f;
    wt[n * kpad + k] = (_Float16)v;
  }
  for (int n = blockIdx.x * blockDim.x + threadIdx.x; n < npad; n += stride)
    bout[n] = (n < d1) ? bsrc[n] : 0.0f;
}

__global__ void zero_kernel(float* __restrict__ p, int n) {
  const int stride = gridDim.x * blockDim.x;
  for (int i = blockIdx.x * blockDim.x + threadIdx.x; i < n; i += stride)
    p[i] = 0.0f;
}

// ---- the wavefront kernel: one workgroup, 8 waves -------------------------
__global__ void __launch_bounds__(256)
wavefront_kernel(const float* __restrict__ X,          // (3,256,256)
                 const _Float16* __restrict__ gWt,
                 const float* __restrict__ gB,
                 float* O)                              // (253,256,12)
{
  extern __shared__ char smem[];
  _Float16* sWt  = (_Float16*)smem;                    // 36352 halves
  float*    sB   = (float*)(smem + BIAS_OFF);          // 832 f32
  _Float16* sAct = (_Float16*)(smem + OGRID_OFF);      // 8 waves x 16 x 64 f16

  const int tid = threadIdx.x;

  // stage all weights + biases into LDS once
  {
    const uint32_t* src = (const uint32_t*)gWt;
    uint32_t* dst = (uint32_t*)sWt;
    for (int i = tid; i < WT_BYTES / 4; i += 256) dst[i] = src[i];
    for (int i = tid; i < BIAS_FLOATS; i += 256) sB[i] = gB[i];
  }
  __syncthreads();

  const int wave = tid >> 5;
  const int lane = tid & 31;
  const int m16  = lane & 15;
  const int hi   = lane >> 4;
  _Float16* act  = sAct + wave * (16 * 64);

  for (int t = 0; t <= TMAX; ++t) {
    const int rmin  = (t >= 251) ? ((t - 250) >> 1) : 0;
    const int rmax  = (251 < (t >> 1)) ? 251 : (t >> 1);
    const int count = rmax - rmin + 1;

    run_mlp<0, 24>(X + 0 * IMG_H * IMG_W, O, sWt, sB, act,
                   t, rmin, count, wave, lane, m16, hi);
    run_mlp<1, 48>(X + 1 * IMG_H * IMG_W, O, sWt, sB, act,
                   t, rmin, count, wave, lane, m16, hi);
    run_mlp<2, 24>(X + 2 * IMG_H * IMG_W, O, sWt, sB, act,
                   t, rmin, count, wave, lane, m16, hi);

    __threadfence_block();
    __syncthreads();
  }
}

// ---- deterministic reduction: 255*sqrt(sum(resid^2)/(3*255*256)) ----------
__global__ void reduce_kernel(const float* __restrict__ O, float* __restrict__ out) {
  __shared__ double sred[256];
  const int tid = threadIdx.x;
  double s = 0.0;
  const int total = 252 * 256 * 3;
  for (int e = tid; e < total; e += 256) {
    const int r   = e / 768;
    const int rem = e - r * 768;
    const int cc  = rem / 3;
    const int ch  = (rem - cc * 3) * 4 + 3;   // channels 3,7,11
    const float v = O[(r + 1) * OSTRIDE + cc * 12 + ch];
    s += (double)v * (double)v;
  }
  sred[tid] = s;
  __syncthreads();
  for (int off = 128; off > 0; off >>= 1) {
    if (tid < off) sred[tid] += sred[tid + off];
    __syncthreads();
  }
  if (tid == 0) {
    const double mse = sred[0] / (3.0 * 255.0 * 256.0);
    out[0] = (float)(255.0 * sqrt(mse));
  }
}

// ---------------------------------------------------------------------------
extern "C" void kernel_launch(void* const* d_in, const int* in_sizes, int n_in,
                              void* d_out, int out_size, void* d_ws, size_t ws_size,
                              hipStream_t stream) {
  (void)in_sizes; (void)n_in; (void)out_size; (void)ws_size;

  const float* X = (const float*)d_in[0];
  char* ws = (char*)d_ws;
  _Float16* wt  = (_Float16*)ws;
  float* bias   = (float*)(ws + BIAS_OFF);
  float* Ogrid  = (float*)(ws + OGRID_OFF);

  // zero output grid (boundary columns/rows must be 0 every call)
  const int On = OROWS * 256 * 12;
  zero_kernel<<<(On + 255) / 256, 256, 0, stream>>>(Ogrid, On);

  // pack + transpose + f16-convert all 24 weight matrices and biases
  for (int mlp = 0; mlp < 3; ++mlp) {
    for (int l = 0; l < 8; ++l) {
      const float* Wp = (const float*)d_in[1 + (mlp * 8 + l) * 2];
      const float* bp = (const float*)d_in[1 + (mlp * 8 + l) * 2 + 1];
      const int d0 = hDIN[mlp][l], d1 = hDOUT[mlp][l];
      const int kpad = (d0 <= 32) ? 32 : 64;
      const int npad = (d1 == 4) ? 16 : ((d1 == 24) ? 32 : 48);
      const int total = npad * kpad;
      prep_weights_kernel<<<(total + 255) / 256, 256, 0, stream>>>(
          Wp, bp, wt + WOFFc[mlp][l], bias + BOFFc[mlp][l], d0, d1, kpad, npad);
    }
  }

  // single persistent workgroup wavefront (8 waves, ~92KB dynamic LDS)
  wavefront_kernel<<<1, 256, SMEM_BYTES, stream>>>(X, wt, bias, Ogrid);

  // deterministic scalar reduction
  reduce_kernel<<<1, 256, 0, stream>>>(Ogrid, (float*)d_out);
}
